// ODEnet_8787503087646
// MI455X (gfx1250) — matrix operand order repaired and checked by reading, not compiled, not run on hardware
//
#include <hip/hip_runtime.h>

// ---------------------------------------------------------------------------
// Fully-fused ODEnet integrator for MI455X (gfx1250, wave32, WMMA).
//  kernel 1 (prep): fp32 weights -> bf16, zero-padded, WMMA-A-friendly layout in d_ws
//  kernel 2 (main): 512 blocks (one per batch item) x 512 threads (16 waves);
//                   runs all n_steps time steps, all 2 nets x 13 GEMM-layers per step,
//                   activations resident in LDS, matmuls via v_wmma_f32_16x16x32_bf16.
// ---------------------------------------------------------------------------

#define AS     266          // LDS activation row stride (bf16 elems); 133 dwords (odd -> no bank conflicts)
#define HB     134          // column base per half (data_l at 0, data_r at 134); col(t) = h*HB + 2 + t
#define NWNET  47616        // padded bf16 weight elems per net
#define NBNET  640          // padded fp32 bias elems per net

typedef __attribute__((ext_vector_type(16))) __bf16   v16bf;
typedef __attribute__((ext_vector_type(2)))  __bf16   v2bf;
typedef __attribute__((ext_vector_type(8)))  float    v8f;
typedef __attribute__((ext_vector_type(4)))  unsigned uv4;
typedef __attribute__((ext_vector_type(4)))  float    fv4;

// Per-layer tables (16 GEMM tensors per net, leaf order = 2*i (w), 2*i+1 (b))
__constant__ int C_OUT[16]  = {16,16,16,16,16,32,32,32,64,64,64,64,64,64,64, 1};
__constant__ int C_OUTP[16] = {16,16,16,16,16,32,32,32,64,64,64,64,64,64,64,16};
__constant__ int C_INC[16]  = { 2,16, 2,16,16,16,32,16,32,64,32,64,64,64,64,64};
__constant__ int C_KP[16]   = {32,32,32,32,32,32,32,32,32,64,32,64,64,64,64,64};
__constant__ int C_K[16]    = { 3, 1, 1, 3, 1, 3, 1, 1, 3, 1, 1, 3, 1, 1, 1, 1};
__constant__ int C_WOFF[16] = {0,1536,2048,2560,4096,4608,7680,8704,9728,15872,19968,22016,34304,38400,42496,46592};
__constant__ int C_BOFF[16] = {0,16,32,48,64,80,112,144,176,240,304,368,432,496,560,624};

// compile-time copies for the main kernel (literal indices only)
#define W_0 0
#define W_1 1536
#define W_2 2048
#define W_3 2560
#define W_4 4096
#define W_5 4608
#define W_6 7680
#define W_7 8704
#define W_8 9728
#define W_9 15872
#define W_10 19968
#define W_11 22016
#define W_12 34304
#define W_13 38400
#define W_14 42496
#define W_15 46592
#define B_0 0
#define B_1 16
#define B_2 32
#define B_3 48
#define B_4 64
#define B_5 80
#define B_6 112
#define B_7 144
#define B_8 176
#define B_9 240
#define B_10 304
#define B_11 368
#define B_12 432
#define B_13 496
#define B_14 560
#define B_15 624

__device__ __forceinline__ __bf16 f2bf(float f) {
    union { float f; unsigned u; } a; a.f = f;
    unsigned r = a.u + 0x7fffu + ((a.u >> 16) & 1u);       // round-to-nearest-even
    union { unsigned short s; __bf16 b; } o; o.s = (unsigned short)(r >> 16);
    return o.b;
}
__device__ __forceinline__ float bf2f(__bf16 b) {
    union { unsigned short s; __bf16 b; } i; i.b = b;
    union { unsigned u; float f; } o; o.u = ((unsigned)i.s) << 16;
    return o.f;
}
// two floats -> packed {lo,hi} bf16 in one u32 (single v_cvt_pk_bf16_f32 when available)
__device__ __forceinline__ unsigned pk2bf(float lo, float hi) {
#if __has_builtin(__builtin_amdgcn_cvt_pk_bf16_f32)
    union { v2bf v; unsigned u; } c;
    c.v = __builtin_amdgcn_cvt_pk_bf16_f32(lo, hi);
    return c.u;
#else
    union { float f; unsigned u; } a, b; a.f = lo; b.f = hi;
    unsigned ra = a.u + 0x7fffu + ((a.u >> 16) & 1u);
    unsigned rb = b.u + 0x7fffu + ((b.u >> 16) & 1u);
    return (ra >> 16) | (rb & 0xffff0000u);
#endif
}
// relu as a single v_med3_f32
__device__ __forceinline__ float relu_f(float x) {
    return __builtin_amdgcn_fmed3f(x, 0.f, __builtin_inff());
}

// ---------------------------------------------------------------------------
// prep kernel: repack fp32 weights -> bf16 [dk][coP][KP] blobs + padded fp32 biases
// ---------------------------------------------------------------------------
struct PtrTab { const float* p[64]; };

__global__ void prep_kernel(PtrTab tab, __bf16* __restrict__ wout, float* __restrict__ bout) {
    const int tens = blockIdx.y;              // 0..31  (net*16 + layer)
    const int net  = tens >> 4, li = tens & 15;
    const float* w = tab.p[net * 32 + 2 * li];
    const float* b = tab.p[net * 32 + 2 * li + 1];
    const int Cout = C_OUT[li], CoutP = C_OUTP[li], Cin = C_INC[li], KP = C_KP[li], k = C_K[li];
    const int n = k * CoutP * KP;
    __bf16* wo = wout + net * NWNET + C_WOFF[li];
    for (int i = blockIdx.x * blockDim.x + threadIdx.x; i < n; i += blockDim.x * gridDim.x) {
        int dk  = i / (CoutP * KP);
        int rem = i - dk * (CoutP * KP);
        int co  = rem / KP;
        int ci  = rem - co * KP;
        float v = (co < Cout && ci < Cin) ? w[(co * Cin + ci) * k + dk] : 0.f;
        wo[i] = f2bf(v);
    }
    if (blockIdx.x == 0 && (int)threadIdx.x < CoutP) {
        int co = threadIdx.x;
        bout[net * NBNET + C_BOFF[li] + co] = (co < Cout) ? b[co] : 0.f;
    }
}

// ---------------------------------------------------------------------------
// WMMA fragment loads
// ---------------------------------------------------------------------------
// B fragment (32x16, K x N) at column offset q: lane col = N = lane%16, K in table order (8h+s, 16+8h+s-8)
__device__ __forceinline__ v16bf load_bfrag_even(const __bf16* row, int q) {
    union { v16bf v; unsigned short u[16]; } f;
    const unsigned short* p = (const unsigned short*)row;
    const int l = threadIdx.x & 31, h = l >> 4, n = l & 15;
    #pragma unroll
    for (int s = 0; s < 16; ++s) {
        int k = (s < 8) ? (8 * h + s) : (16 + 8 * h + (s - 8));
        f.u[s] = p[k * AS + q + n];
    }
    return f.v;
}
// A fragment (16x32, M x K): lane row = M = lane%16, K runs [ci0..ci0+7],[ci0+16..ci0+23]
__device__ __forceinline__ v16bf load_afrag(const __bf16* wrow, int ci0) {
    union { v16bf v; uv4 q[2]; } f;
    f.q[0] = *(const uv4*)(wrow + ci0);
    f.q[1] = *(const uv4*)(wrow + ci0 + 16);
    return f.v;
}
// bias vector for one row tile: two aligned b128 loads seed the whole accumulator
__device__ __forceinline__ v8f load_biasvec(const float* bp) {
    union { v8f v; fv4 q[2]; } b;
    b.q[0] = *(const fv4*)(bp);
    b.q[1] = *(const fv4*)(bp + 4);
    return b.v;
}

// ---------------------------------------------------------------------------
// one fused GEMM layer:  dst = act( W(src,k taps) [+ W2(src2)] [+ src_id] + bias )
//   MT = Cout/16 row tiles, KT = taps (1|3), KPA/KPB = padded K extents
//   dst==nullptr -> final 1x1 conv, write row 0 to outv[2][120], no relu
// ---------------------------------------------------------------------------
template<int MT, int KT, int KPA, int KPB>
__device__ __forceinline__ void gemm_layer(
    __bf16* dst, const __bf16* src,
    const __bf16* gW, const float* gB,
    const __bf16* gW2, const __bf16* src2, const float* gB2,
    const __bf16* src_id, float* outv)
{
    const int tid  = threadIdx.x;
    const int wv   = tid >> 5;          // 16 waves: wave w owns column tile w
    const int lane = tid & 31;
    const int hi   = lane >> 4;
    const int ln   = lane & 15;
    const int hh   = wv >> 3;           // 0 = data_l half, 1 = data_r half
    const int t0   = (wv & 7) << 4;     // column tile start within half
    const int hb   = hh * HB;

    v8f c[MT];
    #pragma unroll
    for (int mt = 0; mt < MT; ++mt) {
        v8f b = load_biasvec(gB + mt * 16 + 8 * hi);
        if (gB2) b = b + load_biasvec(gB2 + mt * 16 + 8 * hi);
        c[mt] = b;
    }

    #pragma unroll
    for (int kc = 0; kc < (KPA >> 5); ++kc) {
        const __bf16* srow = src + (kc * 32) * AS;
        if (KT == 3) {
            #pragma unroll
            for (int dk = 0; dk < 3; ++dk) {
                union { v16bf v; unsigned u[8]; } f;
                f.v = load_bfrag_even(srow, hb + t0 + 1 + dk);
                #pragma unroll
                for (int mt = 0; mt < MT; ++mt) {
                    v16bf a = load_afrag(gW + (dk * (MT * 16) + mt * 16 + ln) * KPA, kc * 32 + 8 * hi);
                    c[mt] = __builtin_amdgcn_wmma_f32_16x16x32_bf16(
                                false, a, false, f.v, (short)0, c[mt], false, false);
                }
            }
        } else {
            v16bf b = load_bfrag_even(srow, hb + 2 + t0);
            #pragma unroll
            for (int mt = 0; mt < MT; ++mt) {
                v16bf a = load_afrag(gW + (mt * 16 + ln) * KPA, kc * 32 + 8 * hi);
                c[mt] = __builtin_amdgcn_wmma_f32_16x16x32_bf16(
                            false, a, false, b, (short)0, c[mt], false, false);
            }
        }
    }

    if (gW2) {   // fused 1x1 shortcut conv accumulating into the same C tile
        #pragma unroll
        for (int kc = 0; kc < (KPB >> 5); ++kc) {
            const __bf16* srow = src2 + (kc * 32) * AS;
            v16bf b = load_bfrag_even(srow, hb + 2 + t0);
            #pragma unroll
            for (int mt = 0; mt < MT; ++mt) {
                v16bf a = load_afrag(gW2 + (mt * 16 + ln) * KPB, kc * 32 + 8 * hi);
                c[mt] = __builtin_amdgcn_wmma_f32_16x16x32_bf16(
                            false, a, false, b, (short)0, c[mt], false, false);
            }
        }
    }

    if (src_id) {  // identity residual
        const int col = hb + 2 + t0 + ln;
        #pragma unroll
        for (int mt = 0; mt < MT; ++mt)
            #pragma unroll
            for (int r = 0; r < 8; ++r)
                c[mt][r] += bf2f(src_id[(mt * 16 + 8 * hi + r) * AS + col]);
    }

    __syncthreads();   // all reads of src/src2/dst-aliased buffers done

    if (dst) {
        const int col = hb + 2 + t0 + ln;
        unsigned short hv[MT][8];
        #pragma unroll
        for (int mt = 0; mt < MT; ++mt) {
            #pragma unroll
            for (int r = 0; r < 8; r += 2) {               // relu(med3) + packed bf16 cvt
                unsigned pk = pk2bf(relu_f(c[mt][r]), relu_f(c[mt][r + 1]));
                hv[mt][r]     = (unsigned short)(pk & 0xffffu);
                hv[mt][r + 1] = (unsigned short)(pk >> 16);
            }
            #pragma unroll
            for (int r = 0; r < 8; ++r)
                *(unsigned short*)(dst + (mt * 16 + 8 * hi + r) * AS + col) = hv[mt][r];
        }
        if (t0 == 0 && ln == 0) {                         // left edge-pad clone (one exec toggle)
            #pragma unroll
            for (int mt = 0; mt < MT; ++mt)
                #pragma unroll
                for (int r = 0; r < 8; ++r)
                    *(unsigned short*)(dst + (mt * 16 + 8 * hi + r) * AS + col - 1) = hv[mt][r];
        }
        if (t0 == 112 && ln == 7) {                       // right edge-pad clone
            #pragma unroll
            for (int mt = 0; mt < MT; ++mt)
                #pragma unroll
                for (int r = 0; r < 8; ++r)
                    *(unsigned short*)(dst + (mt * 16 + 8 * hi + r) * AS + col + 1) = hv[mt][r];
        }
    } else if (outv) {
        if (hi == 0) {                    // M = 0 lives in VGPR0 of lanes 0..15
            int t = t0 + ln;
            if (t < 120) outv[hh * 120 + t] = c[0][0];
        }
    }
    __syncthreads();
}

// ---------------------------------------------------------------------------
// main kernel: one block per batch item, full time loop in-kernel
// ---------------------------------------------------------------------------
__global__ __launch_bounds__(512) void ode_main(
    const float* __restrict__ z0, const __bf16* __restrict__ wsW,
    const float* __restrict__ wsB, const int* __restrict__ t1p,
    float* __restrict__ out)
{
    __shared__ __bf16 P0[64 * AS];
    __shared__ __bf16 P1[64 * AS];
    __shared__ float  zb[120], zn[120], lamv[240], Lv[240];

    const int tid = threadIdx.x;
    const int nb  = blockIdx.x;

    for (int i = tid; i < 64 * AS; i += blockDim.x) { P0[i] = f2bf(0.f); P1[i] = f2bf(0.f); }
    if (tid < 120) zb[tid] = z0[nb * 120 + tid];
    const int t1 = *t1p;
    int nsteps = t1 * 4;                 // round(t1 / 0.25)
    if (nsteps < 1) nsteps = 1;
    const float hstep = (float)t1 / (float)nsteps;
    __syncthreads();

    for (int s = 0; s < nsteps; ++s) {
        for (int net = 0; net < 2; ++net) {
            const __bf16* WB = wsW + net * NWNET;
            const float*  BB = wsB + net * NBNET;

            // warm WGP$ with this net's weight blob (global_prefetch_b8)
            for (int off = tid * 128; off < NWNET * 2; off += 512 * 128)
                __builtin_prefetch((const char*)WB + off, 0, 3);

            // stage data_l (ch0=ul, ch1=z) and data_r (ch0=z, ch1=ur) into P0 rows 0..1
            if (tid < 240) {
                int hh = tid / 120, t = tid - hh * 120;
                float um  = zb[t];
                float ulv = (t == 0)   ? zb[0]   : zb[t - 1];
                float urv = (t == 119) ? zb[119] : zb[t + 1];
                float r0 = hh == 0 ? ulv : um;
                float r1 = hh == 0 ? um  : urv;
                int col = hh * HB + 2 + t;
                __bf16 b0 = f2bf(r0), b1 = f2bf(r1);
                P0[0 * AS + col] = b0;  P0[1 * AS + col] = b1;
                if (t == 0)   { P0[0 * AS + col - 1] = b0; P0[1 * AS + col - 1] = b1; }
                if (t == 119) { P0[0 * AS + col + 1] = b0; P0[1 * AS + col + 1] = b1; }
            }
            __syncthreads();

            float* ov = net ? Lv : lamv;
            const __bf16* NB0 = nullptr; const float* NF = nullptr;

            // block1: 2->16, k3 conv + 1x1 + 1x1 shortcut
            gemm_layer<1,3,32,32>(P1, P0, WB+W_0,  BB+B_0,  NB0, NB0, NF, NB0, nullptr);
            gemm_layer<1,1,32,32>(P0, P1, WB+W_1,  BB+B_1,  WB+W_2,  P0, BB+B_2,  NB0, nullptr);
            // block2: 16->16, identity residual
            gemm_layer<1,3,32,32>(P1, P0, WB+W_3,  BB+B_3,  NB0, NB0, NF, NB0, nullptr);
            gemm_layer<1,1,32,32>(P0, P1, WB+W_4,  BB+B_4,  NB0, NB0, NF, P0,  nullptr);
            // block3: 16->32
            gemm_layer<2,3,32,32>(P1, P0, WB+W_5,  BB+B_5,  NB0, NB0, NF, NB0, nullptr);
            gemm_layer<2,1,32,32>(P0, P1, WB+W_6,  BB+B_6,  WB+W_7,  P0, BB+B_7,  NB0, nullptr);
            // block4: 32->64
            gemm_layer<4,3,32,32>(P1, P0, WB+W_8,  BB+B_8,  NB0, NB0, NF, NB0, nullptr);
            gemm_layer<4,1,64,32>(P0, P1, WB+W_9,  BB+B_9,  WB+W_10, P0, BB+B_10, NB0, nullptr);
            // block5: 64->64, identity
            gemm_layer<4,3,64,32>(P1, P0, WB+W_11, BB+B_11, NB0, NB0, NF, NB0, nullptr);
            gemm_layer<4,1,64,32>(P0, P1, WB+W_12, BB+B_12, NB0, NB0, NF, P0,  nullptr);
            // block6: 64->64 (k=1), identity
            gemm_layer<4,1,64,32>(P1, P0, WB+W_13, BB+B_13, NB0, NB0, NF, NB0, nullptr);
            gemm_layer<4,1,64,32>(P0, P1, WB+W_14, BB+B_14, NB0, NB0, NF, P0,  nullptr);
            // final 64->1 (padded to 16 rows), no relu, write to lamv / Lv
            gemm_layer<1,1,64,32>(nullptr, P0, WB+W_15, BB+B_15, NB0, NB0, NF, NB0, ov);
        }

        // flux + Euler update
        if (tid < 120) {
            int t = tid;
            float um  = zb[t];
            float ulv = (t == 0)   ? zb[0]   : zb[t - 1];
            float urv = (t == 119) ? zb[119] : zb[t + 1];
            float ll = lamv[t] * 0.1f,  lr = lamv[120 + t] * 0.1f;
            float Ll = Lv[t],           Lr = Lv[120 + t];
            float Rl = Ll / (Ll * Ll),  Rr = Lr / (Lr * Lr);
            float o  = Rr * (lr - fabsf(lr)) * Lr * (urv - um)
                     + Rl * (ll + fabsf(ll)) * Ll * (um  - ulv);
            float du = -(o / (2.0f * 0.01f));
            zn[t] = um + hstep * du;
        }
        __syncthreads();
        // cal_boundry: [u[100:110], u[10:110], u[10:20]]
        if (tid < 120) {
            int t = tid;
            zb[t] = (t < 10) ? zn[100 + t] : ((t < 110) ? zn[t] : zn[t - 100]);
        }
        __syncthreads();
    }

    if (tid < 120) out[nb * 120 + tid] = zb[tid];
}

// ---------------------------------------------------------------------------
extern "C" void kernel_launch(void* const* d_in, const int* in_sizes, int n_in,
                              void* d_out, int out_size, void* d_ws, size_t ws_size,
                              hipStream_t stream) {
    (void)in_sizes; (void)out_size; (void)ws_size;
    // leaves: [0]=z0, [1..32]=lam_params, [33..64]=L_params, [last]=t1_t0
    PtrTab tab;
    for (int i = 0; i < 64; ++i) tab.p[i] = (const float*)d_in[1 + i];

    __bf16* wsW = (__bf16*)d_ws;
    float*  wsB = (float*)((char*)d_ws + (size_t)2 * NWNET * sizeof(__bf16));

    dim3 gp(8, 32, 1);
    prep_kernel<<<gp, 256, 0, stream>>>(tab, wsW, wsB);

    const int* t1p = (const int*)d_in[n_in - 1];
    ode_main<<<512, 512, 0, stream>>>((const float*)d_in[0], wsW, wsB, t1p, (float*)d_out);
}
